// MeshGraphBlock_67851893342549
// MI455X (gfx1250) — matrix operand order repaired and checked by reading, not compiled, not run on hardware
//
#include <hip/hip_runtime.h>
#include <hip/hip_bf16.h>
#include <stdint.h>
#include <math.h>

// ---------------- compile-time problem constants (D=128 from reference) ----
constexpr int   DIM    = 128;     // D
constexpr int   TWOD   = 256;     // 2D
constexpr int   TILE_M = 16;      // node rows per tile
constexpr int   HSTR   = 272;     // LDS row stride in halves (544B = 16*34: b128-aligned, bank-conflict padded)
constexpr int   XSTR   = 132;     // LDS row stride in floats for residual copy
constexpr int   PERSIST_BLOCKS = 2048;  // persistent workgroups (weights loaded once per block)

typedef __attribute__((ext_vector_type(16))) _Float16 v16h;
typedef __attribute__((ext_vector_type(8)))  _Float16 v8h;
typedef __attribute__((ext_vector_type(8)))  float    v8f;

// ---------------------------------------------------------------- helpers --
__device__ __forceinline__ float waveSum32(float v) {
#pragma unroll
    for (int o = 16; o >= 1; o >>= 1) v += __shfl_xor(v, o, 32);
    return v;
}

__device__ __forceinline__ v16h ld_frag(const _Float16* p0, const _Float16* p1) {
    // two 16B LDS loads -> one 16-half fragment
    v8h lo = *(const v8h*)p0;
    v8h hi = *(const v8h*)p1;
    return __builtin_shufflevector(lo, hi, 0,1,2,3,4,5,6,7,8,9,10,11,12,13,14,15);
}

// ---------------------------------------------------- weight f32 -> f16^T --
__global__ void cvt_weights(const float* __restrict__ W1, const float* __restrict__ W2,
                            _Float16* __restrict__ w1t, _Float16* __restrict__ w2t) {
    int i = blockIdx.x * blockDim.x + threadIdx.x;
    if (i < TWOD * TWOD) {                 // W1 [k][n] -> w1t [n][k]
        int k = i / TWOD, n = i % TWOD;
        w1t[(size_t)n * TWOD + k] = (_Float16)W1[i];
    }
    if (i < TWOD * DIM) {                  // W2 [k][n] -> w2t [n][k]
        int k = i / DIM, n = i % DIM;
        w2t[(size_t)n * TWOD + k] = (_Float16)W2[i];
    }
}

// ------------------------------------------------- edge gather/scatter-add --
// one wave per edge; 32 lanes x float4 = 128 dims; bandwidth/atomic bound
__global__ void scatter_edges(const float* __restrict__ x,
                              const int* __restrict__ esrc,
                              const int* __restrict__ edst,
                              float* __restrict__ nb,
                              int E, int N, int Bb) {
    int wave = (blockIdx.x * blockDim.x + threadIdx.x) >> 5;
    int lane = threadIdx.x & 31;
    if (wave >= E) return;
    int s = esrc[wave];
    int d = edst[wave];
    for (int b = 0; b < Bb; ++b) {
        const float4 v = ((const float4*)(x + ((size_t)b * N + s) * DIM))[lane];
        float* o = nb + ((size_t)b * N + d) * DIM + (size_t)lane * 4;
        atomicAdd(o + 0, v.x);
        atomicAdd(o + 1, v.y);
        atomicAdd(o + 2, v.z);
        atomicAdd(o + 3, v.w);
    }
}

// ------------------------------------------------------- fused node kernel --
// Persistent workgroups: each block async-DMAs the full f16 W1^T/W2^T (192KB)
// into LDS ONCE, then grid-strides over 16-row node tiles:
//   LN(x) | LN(neighbor/deg) -> h[16x256] f16 in LDS,
//   GEMM1 (h @ W1 + b1, exact-erf GELU) -> g[16x256] f16,
//   GEMM2 (g @ W2 + b2) + residual x   -> out.
__global__ void __launch_bounds__(256, 1)
fused_node_kernel(const float* __restrict__ x,  const float* __restrict__ nb,
                  const float* __restrict__ deg,
                  const float* __restrict__ sng, const float* __restrict__ snb,
                  const float* __restrict__ nng, const float* __restrict__ nnb,
                  const _Float16* __restrict__ w1t, const _Float16* __restrict__ w2t,
                  const float* __restrict__ b1, const float* __restrict__ b2,
                  float* __restrict__ out, long R, int N, long numTiles) {
    extern __shared__ __align__(16) char smem[];
    _Float16* sW1 = (_Float16*)smem;                 // [256][256] halves (n-major)
    _Float16* sW2 = sW1 + TWOD * TWOD;               // [128][256] halves (n-major)
    _Float16* sH  = sW2 + TWOD * DIM;                // [16][HSTR]
    _Float16* sG  = sH + TILE_M * HSTR;              // [16][HSTR]
    float*    sX  = (float*)(sG + TILE_M * HSTR);    // [16][XSTR] residual
    float*    sB1 = sX + TILE_M * XSTR;              // [256]
    float*    sB2 = sB1 + TWOD;                      // [128]

    const int tid  = threadIdx.x;
    const int lane = tid & 31;
    const int w    = tid >> 5;

    // ---- one-time: async bulk copy of f16 weights (contiguous W1t|W2t) ----
    {
        const char*  gsrc   = (const char*)w1t;
        unsigned int ldsOff = (unsigned int)(uintptr_t)smem;   // low 32b = LDS addr
        const int nchunks = (TWOD * TWOD + TWOD * DIM) * 2 / 16;   // 12288 x 16B
        for (int i = tid; i < nchunks; i += 256) {
            unsigned int       loff = ldsOff + (unsigned int)i * 16u;
            unsigned long long ga   = (unsigned long long)(uintptr_t)(gsrc + (size_t)i * 16);
            asm volatile("global_load_async_to_lds_b128 %0, %1, off"
                         :: "v"(loff), "v"(ga) : "memory");
        }
    }
    if (tid < TWOD) sB1[tid] = b1[tid];
    if (tid < DIM)  sB2[tid] = b2[tid];

    // LN params (per-lane float4 slices), loop-invariant
    const float4 gx = ((const float4*)sng)[lane];
    const float4 bx = ((const float4*)snb)[lane];
    const float4 gn = ((const float4*)nng)[lane];
    const float4 bn = ((const float4*)nnb)[lane];

    // WMMA fragment lane mapping (wave32, 16-bit A 16x32 / B 32x16)
    const int m  = lane & 15;            // A row / B col / C col within tile
    const int kb = (lane >> 4) << 3;     // A: half-wave K base (0/8), +16 for hi half
    const int kh = (lane >> 4) << 4;     // unused; B also uses kb (0/8, +16 for hi half)
    const int rj = (lane >> 4) << 3;     // C/D: row base for VGPR j (0/8)

    asm volatile("s_wait_asynccnt 0x0" ::: "memory");   // weights resident in LDS
    __syncthreads();

    // ================== persistent grid-stride loop over row tiles =========
    for (long tile = blockIdx.x; tile < numTiles; tile += gridDim.x) {
        const long tileBase = tile * TILE_M;

        // ---- LayerNorm phase: 2 rows per wave ----------------------------
#pragma unroll
        for (int pass = 0; pass < 2; ++pass) {
            const int row = w * 2 + pass;
            long gr  = tileBase + row;
            long grc = gr < R ? gr : (R - 1);
            // prefetch this block's NEXT tile rows (global_prefetch_b8)
            long nxt = gr + (long)gridDim.x * TILE_M;
            const float* pf = x + (size_t)((nxt < R) ? nxt : grc) * DIM;
            __builtin_prefetch(pf, 0, 1);

            float4 xv = ((const float4*)(x + (size_t)grc * DIM))[lane];
            float* xr = sX + row * XSTR + lane * 4;
            xr[0] = xv.x; xr[1] = xv.y; xr[2] = xv.z; xr[3] = xv.w;

            // LN(x)
            float mean = waveSum32(xv.x + xv.y + xv.z + xv.w) * (1.0f / DIM);
            float d0 = xv.x - mean, d1 = xv.y - mean, d2 = xv.z - mean, d3 = xv.w - mean;
            float var  = waveSum32(d0*d0 + d1*d1 + d2*d2 + d3*d3) * (1.0f / DIM);
            float rstd = rsqrtf(var + 1e-5f);
            _Float16* hr = sH + row * HSTR + lane * 4;
            hr[0] = (_Float16)(d0 * rstd * gx.x + bx.x);
            hr[1] = (_Float16)(d1 * rstd * gx.y + bx.y);
            hr[2] = (_Float16)(d2 * rstd * gx.z + bx.z);
            hr[3] = (_Float16)(d3 * rstd * gx.w + bx.w);

            // LN(neighbor / max(deg,1))
            int   node = (int)(grc % N);
            float inv  = 1.0f / fmaxf(deg[node], 1.0f);
            float4 nv = ((const float4*)(nb + (size_t)grc * DIM))[lane];
            nv.x *= inv; nv.y *= inv; nv.z *= inv; nv.w *= inv;
            mean = waveSum32(nv.x + nv.y + nv.z + nv.w) * (1.0f / DIM);
            d0 = nv.x - mean; d1 = nv.y - mean; d2 = nv.z - mean; d3 = nv.w - mean;
            var  = waveSum32(d0*d0 + d1*d1 + d2*d2 + d3*d3) * (1.0f / DIM);
            rstd = rsqrtf(var + 1e-5f);
            _Float16* hn = sH + row * HSTR + DIM + lane * 4;
            hn[0] = (_Float16)(d0 * rstd * gn.x + bn.x);
            hn[1] = (_Float16)(d1 * rstd * gn.y + bn.y);
            hn[2] = (_Float16)(d2 * rstd * gn.z + bn.z);
            hn[3] = (_Float16)(d3 * rstd * gn.w + bn.w);
        }
        __syncthreads();

        // ---- GEMM1: h[16x256] @ W1[256x256], 8 waves x 2 column tiles ----
        v8f acc1[2];
#pragma unroll
        for (int t = 0; t < 2; ++t) {
            const int n0 = w * 32 + t * 16;
            v8f acc = {};
#pragma unroll
            for (int ks = 0; ks < 8; ++ks) {
                const int k0 = ks * 32;
                const _Float16* pa = sH  + m * HSTR + k0 + kb;
                const _Float16* pb = sW1 + (n0 + m) * TWOD + k0 + kb;
                v16h a  = ld_frag(pa, pa + 16);
                v16h bm = ld_frag(pb, pb + 16);
                acc = __builtin_amdgcn_wmma_f32_16x16x32_f16(
                          false, a, false, bm, (short)0, acc, false, false);
            }
            acc1[t] = acc;
        }
        // bias + exact-erf GELU -> g (f16)
#pragma unroll
        for (int t = 0; t < 2; ++t) {
            const int n0 = w * 32 + t * 16;
            const float bias = sB1[n0 + m];
#pragma unroll
            for (int j = 0; j < 8; ++j) {
                float vv = acc1[t][j] + bias;
                float ge = 0.5f * vv * (1.0f + erff(vv * 0.70710678118654752f));
                sG[(j + rj) * HSTR + n0 + m] = (_Float16)ge;
            }
        }
        __syncthreads();

        // ---- GEMM2: g[16x256] @ W2[256x128], 8 waves x 1 column tile -----
        {
            const int n0 = w * 16;
            v8f acc = {};
#pragma unroll
            for (int ks = 0; ks < 8; ++ks) {
                const int k0 = ks * 32;
                const _Float16* pa = sG  + m * HSTR + k0 + kb;
                const _Float16* pb = sW2 + (n0 + m) * TWOD + k0 + kb;
                v16h a  = ld_frag(pa, pa + 16);
                v16h bm = ld_frag(pb, pb + 16);
                acc = __builtin_amdgcn_wmma_f32_16x16x32_f16(
                          false, a, false, bm, (short)0, acc, false, false);
            }
            const float bias = sB2[n0 + m];
#pragma unroll
            for (int j = 0; j < 8; ++j) {
                const int row = j + rj;
                long gr = tileBase + row;
                if (gr < R)
                    out[(size_t)gr * DIM + n0 + m] =
                        sX[row * XSTR + n0 + m] + acc[j] + bias;
            }
        }
        __syncthreads();   // protect sH/sG/sX before next iteration's LN writes
    }
}

// --------------------------------------------------------------- launcher --
extern "C" void kernel_launch(void* const* d_in, const int* in_sizes, int n_in,
                              void* d_out, int out_size, void* d_ws, size_t ws_size,
                              hipStream_t stream) {
    const float* x    = (const float*)d_in[0];
    const int*   esrc = (const int*)  d_in[1];
    const int*   edst = (const int*)  d_in[2];
    const float* deg  = (const float*)d_in[3];
    const float* sng  = (const float*)d_in[4];
    const float* snb  = (const float*)d_in[5];
    const float* nng  = (const float*)d_in[6];
    const float* nnb  = (const float*)d_in[7];
    const float* W1   = (const float*)d_in[8];
    const float* b1   = (const float*)d_in[9];
    const float* W2   = (const float*)d_in[10];
    const float* b2   = (const float*)d_in[11];

    const int  E     = in_sizes[1];
    const int  N     = in_sizes[3];
    const long total = (long)in_sizes[0];           // B*N*D
    const int  Bb    = (int)(total / ((long)N * DIM));
    const long R     = (long)Bb * N;
    const long numTiles = (R + TILE_M - 1) / TILE_M;

    // workspace layout: [neighbor f32 B*N*D][w1t f16 256*256][w2t f16 128*256]
    float*    nbuf = (float*)d_ws;
    _Float16* w1t  = (_Float16*)((char*)d_ws + (size_t)total * sizeof(float));
    _Float16* w2t  = w1t + (size_t)TWOD * TWOD;

    hipMemsetAsync(d_ws, 0, (size_t)total * sizeof(float), stream);

    cvt_weights<<<(TWOD * TWOD + 255) / 256, 256, 0, stream>>>(W1, W2, w1t, w2t);

    scatter_edges<<<(E + 7) / 8, 256, 0, stream>>>(x, esrc, edst, nbuf, E, N, Bb);

    const size_t shmem =
        (size_t)(TWOD * TWOD + TWOD * DIM + 2 * TILE_M * HSTR) * sizeof(_Float16) +
        (size_t)(TILE_M * XSTR + TWOD + DIM) * sizeof(float);   // ~224 KB (<320 KB/WGP)

    const long grid = numTiles < PERSIST_BLOCKS ? numTiles : PERSIST_BLOCKS;
    fused_node_kernel<<<(int)grid, 256, shmem, stream>>>(
        x, nbuf, deg, sng, snb, nng, nnb, w1t, w2t, b1, b2,
        (float*)d_out, R, N, numTiles);
}
